// Decoder_25013889532485
// MI455X (gfx1250) — matrix operand order repaired and checked by reading, not compiled, not run on hardware
//
#include <hip/hip_runtime.h>
#include <hip/hip_bf16.h>

// ---------------------------------------------------------------------------
// DA-RNN decoder on gfx1250: bf16 WMMA for all GEMMs, fp32 accumulate.
// B=512, T-1=127, E=256, D=256.
// All GEMM A/B operands pre-converted to bf16 so the inner loops are pure
// global_load_b128 + v_wmma_f32_16x16x32_bf16 (4 N-tiles per wave, A reuse).
// ---------------------------------------------------------------------------

typedef __attribute__((ext_vector_type(16))) __bf16 v16bf;
typedef __attribute__((ext_vector_type(8)))  __bf16 v8bf;
typedef __attribute__((ext_vector_type(8)))  float  v8f;

#define BATCH 512
#define TSTEPS 127
#define EDIM 256
#define DDIM 256

// float -> bf16 round-to-nearest-even (bit-level; used only in one-time /
// pointwise conversion kernels, never in GEMM inner loops)
static __device__ inline __bf16 f2bf(float x) {
    union { float f; unsigned u; } v; v.f = x;
    unsigned r = v.u + 0x7fffu + ((v.u >> 16) & 1u);
    unsigned short h = (unsigned short)(r >> 16);
    __bf16 out;
    __builtin_memcpy(&out, &h, 2);
    return out;
}

static __device__ inline float wave_sum(float v) {
#pragma unroll
    for (int o = 16; o > 0; o >>= 1) v += __shfl_xor(v, o, 32);
    return v;
}
static __device__ inline float wave_max(float v) {
#pragma unroll
    for (int o = 16; o > 0; o >>= 1) v = fmaxf(v, __shfl_xor(v, o, 32));
    return v;
}

// ---------------------------------------------------------------------------
// 16x64 output strip per wave: one A fragment (16xK, bf16 row-major) reused
// against 4 B tiles (B stored output-column-major: Bbf[n*ldb + k]).
// K must be a multiple of 32. VGPR k order used here (contiguous, same for A and B; wave32):
//   A lanes 0-15: K {0..15};        lanes 16-31: K {16..31}
//   B lanes 0-15: K {0..15};        lanes 16-31: K {16..31}
//   C/D VGPR r  : row m0+r (lanes 0-15) / m0+8+r (lanes 16-31), col = lane%16
// ---------------------------------------------------------------------------
static __device__ inline void wmma_strip4(const __bf16* __restrict__ A, int lda,
                                          const __bf16* __restrict__ Bbf, int ldb,
                                          int K, int m0, int n0, int lane,
                                          v8f acc[4]) {
    const int row   = m0 + (lane & 15);
    const int ahalf = (lane >> 4) << 4;          // 0 or 16
    const int bhalf = (lane >> 4) << 4;          // 0 or 16
    const __bf16* arow = A + (size_t)row * lda;
    const __bf16* brow = Bbf + (size_t)(n0 + (lane & 15)) * ldb;
    for (int k0 = 0; k0 < K; k0 += 32) {
        v16bf a = *(const v16bf*)(arow + k0 + ahalf);       // 32B aligned
#pragma unroll
        for (int j = 0; j < 4; ++j) {
            v16bf b = *(const v16bf*)(brow + (size_t)(j * 16) * ldb + k0 + bhalf);
            acc[j] = __builtin_amdgcn_wmma_f32_16x16x32_bf16(
                         false, a, false, b, (short)0, acc[j], false, false);
        }
    }
}

// ---------------------------------------------------------------------------
// Conversion kernels (one-time)
// ---------------------------------------------------------------------------
__global__ __launch_bounds__(256) void cvt_bf16_flat(__bf16* __restrict__ dst,
                                                     const float* __restrict__ src,
                                                     int n) {
    int idx = blockIdx.x * 256 + threadIdx.x;
    if (idx < n) dst[idx] = f2bf(src[idx]);
}

__global__ __launch_bounds__(256) void cvt_bf16_strided(__bf16* __restrict__ dst,
                                                        const float* __restrict__ src,
                                                        int rows, int rowlen, int stride) {
    int idx = blockIdx.x * 256 + threadIdx.x;
    int n = rows * rowlen;
    if (idx < n) {
        int r = idx / rowlen, cidx = idx - r * rowlen;
        dst[idx] = f2bf(src[(size_t)r * stride + cidx]);
    }
}

// ---------------------------------------------------------------------------
// enc_proj[m][f] = sum_e X[m][e]*W1e[f][e] + attn_b1[f]   (attn_b1 folded in)
// M = 65024 (4064 tiles), N = 256 (4 strip-groups of 64), K = 256.
// ---------------------------------------------------------------------------
__global__ __launch_bounds__(256) void gemm_encproj(const __bf16* __restrict__ Xb,
                                                    const __bf16* __restrict__ W1e,
                                                    const float* __restrict__ b1,
                                                    float* __restrict__ encp) {
    const int lane = threadIdx.x & 31, wid = threadIdx.x >> 5;
    const int job = blockIdx.x * 8 + wid;        // 4064*4 jobs
    const int m0 = (job >> 2) << 4;
    const int n0 = (job & 3) << 6;
    v8f acc[4] = {};
    wmma_strip4(Xb, EDIM, W1e, EDIM, EDIM, m0, n0, lane, acc);
#pragma unroll
    for (int j = 0; j < 4; ++j) {
        const int col = n0 + j * 16 + (lane & 15);
        const float bias = b1[col];
#pragma unroll
        for (int i = 0; i < 8; ++i) {
            const int r = m0 + i + ((lane >> 4) << 3);
            encp[(size_t)r * EDIM + col] = acc[j][i] + bias;
        }
    }
}

// ---------------------------------------------------------------------------
// hc_proj[b][f] = sum_k HC[b][k]*W1hc[f][k]    M=512, N=256, K=512
// ---------------------------------------------------------------------------
__global__ __launch_bounds__(256) void gemm_hcproj(const __bf16* __restrict__ HCb,
                                                   const __bf16* __restrict__ W1hc,
                                                   float* __restrict__ hcp) {
    const int lane = threadIdx.x & 31, wid = threadIdx.x >> 5;
    const int job = blockIdx.x * 8 + wid;        // 32*4 jobs -> 16 blocks
    const int m0 = (job >> 2) << 4;
    const int n0 = (job & 3) << 6;
    v8f acc[4] = {};
    wmma_strip4(HCb, 2 * DDIM, W1hc, 2 * DDIM, 2 * DDIM, m0, n0, lane, acc);
#pragma unroll
    for (int j = 0; j < 4; ++j) {
        const int col = n0 + j * 16 + (lane & 15);
#pragma unroll
        for (int i = 0; i < 8; ++i) {
            const int r = m0 + i + ((lane >> 4) << 3);
            hcp[(size_t)r * EDIM + col] = acc[j][i];
        }
    }
}

// ---------------------------------------------------------------------------
// gates[b][j] = sum_k h[b][k]*Whh[j][k] + yt[b]*Wih[j] + bih[j] + bhh[j]
// M=512, N=1024, K=256.  h = HCb[b][0:256] (lda=512, K-loop stops at 256).
// ---------------------------------------------------------------------------
__global__ __launch_bounds__(256) void gemm_gates(const __bf16* __restrict__ HCb,
                                                  const __bf16* __restrict__ Whh,
                                                  const float* __restrict__ yt,
                                                  const float* __restrict__ Wih,
                                                  const float* __restrict__ bih,
                                                  const float* __restrict__ bhh,
                                                  float* __restrict__ gates) {
    const int lane = threadIdx.x & 31, wid = threadIdx.x >> 5;
    const int job = blockIdx.x * 8 + wid;        // 32*16 jobs -> 64 blocks
    const int m0 = (job >> 4) << 4;
    const int n0 = (job & 15) << 6;
    v8f acc[4] = {};
    wmma_strip4(HCb, 2 * DDIM, Whh, DDIM, DDIM, m0, n0, lane, acc);
#pragma unroll
    for (int j = 0; j < 4; ++j) {
        const int col = n0 + j * 16 + (lane & 15);
        const float wih = Wih[col];
        const float bb = bih[col] + bhh[col];
#pragma unroll
        for (int i = 0; i < 8; ++i) {
            const int r = m0 + i + ((lane >> 4) << 3);
            gates[(size_t)r * 4 * DDIM + col] = acc[j][i] + yt[r] * wih + bb;
        }
    }
}

// ---------------------------------------------------------------------------
// Per-batch fused attention: scores -> softmax -> context -> y_tilde.
// One block (256 threads) per batch element. Wave w handles timesteps
// t = w, w+8, ... ; within a wave, lane l reduces f = l + 32*j.
// ---------------------------------------------------------------------------
__global__ __launch_bounds__(256) void attn_context(const float* __restrict__ hcp,
                                                    const float* __restrict__ encp,
                                                    const float* __restrict__ X,
                                                    const float* __restrict__ W2,
                                                    const float* __restrict__ b2,
                                                    const float* __restrict__ fcW,
                                                    const float* __restrict__ fcb,
                                                    const float* __restrict__ yh,
                                                    int step,
                                                    float* __restrict__ ytilde) {
    __shared__ float hcs[EDIM];
    __shared__ float w2s[EDIM];
    __shared__ float sc[128];
    __shared__ float red[8];

    const int b = blockIdx.x;
    const int tid = threadIdx.x;
    const int lane = tid & 31, wid = tid >> 5;

    hcs[tid] = hcp[(size_t)b * EDIM + tid];
    w2s[tid] = W2[tid];
    __syncthreads();

    const float bias2 = b2[0];
    // ---- scores ----
    for (int t = wid; t < TSTEPS; t += 8) {
        const float* ep = encp + ((size_t)b * TSTEPS + t) * EDIM;
        float v = 0.f;
#pragma unroll
        for (int j = 0; j < 8; ++j) {
            const int f = lane + (j << 5);
            v += tanhf(hcs[f] + ep[f]) * w2s[f];
        }
        v = wave_sum(v);
        if (lane == 0) sc[t] = v + bias2;
    }
    __syncthreads();

    // ---- softmax over t (127 valid) ----
    float x = (tid < TSTEPS) ? sc[tid] : -3.4e38f;
    float m = wave_max(x);
    if (lane == 0) red[wid] = m;
    __syncthreads();
    float mx = red[0];
#pragma unroll
    for (int i = 1; i < 8; ++i) mx = fmaxf(mx, red[i]);
    __syncthreads();

    float e = (tid < TSTEPS) ? __expf(x - mx) : 0.f;
    float s = wave_sum(e);
    if (lane == 0) red[wid] = s;
    __syncthreads();
    float ssum = 0.f;
#pragma unroll
    for (int i = 0; i < 8; ++i) ssum += red[i];
    __syncthreads();
    if (tid < TSTEPS) sc[tid] = e / ssum;
    __syncthreads();

    // ---- context + y_tilde ----
    const float* xb = X + (size_t)b * TSTEPS * EDIM;
    float ctx = 0.f;
    for (int t = 0; t < TSTEPS; ++t) ctx += sc[t] * xb[(size_t)t * EDIM + tid];
    float yp = ctx * fcW[tid];
    float ys = wave_sum(yp);
    if (lane == 0) red[wid] = ys;
    __syncthreads();
    if (tid == 0) {
        float tot = 0.f;
#pragma unroll
        for (int i = 0; i < 8; ++i) tot += red[i];
        ytilde[b] = tot + yh[(size_t)b * TSTEPS + step] * fcW[EDIM] + fcb[0];
    }
}

// ---------------------------------------------------------------------------
// LSTM pointwise: updates fp32 HC buffer (row b: [h(256)|c(256)]) and its
// bf16 mirror HCb used as the WMMA A operand next step.
// ---------------------------------------------------------------------------
static __device__ inline float sigm(float x) { return 1.f / (1.f + __expf(-x)); }

__global__ __launch_bounds__(256) void lstm_pointwise(const float* __restrict__ gates,
                                                      float* __restrict__ HC,
                                                      __bf16* __restrict__ HCb) {
    const int idx = blockIdx.x * 256 + threadIdx.x;   // 512*256
    const int b = idx >> 8, d = idx & 255;
    const float* g = gates + (size_t)b * 4 * DDIM;
    const float gi = g[d];
    const float gf = g[DDIM + d];
    const float gg = g[2 * DDIM + d];
    const float go = g[3 * DDIM + d];
    float* hc = HC + (size_t)b * 2 * DDIM;
    __bf16* hcb = HCb + (size_t)b * 2 * DDIM;
    const float c_old = hc[DDIM + d];
    const float c_new = sigm(gf) * c_old + sigm(gi) * tanhf(gg);
    const float h_new = sigm(go) * tanhf(c_new);
    hc[d] = h_new;
    hc[DDIM + d] = c_new;
    hcb[d] = f2bf(h_new);
    hcb[DDIM + d] = f2bf(c_new);
}

__global__ __launch_bounds__(256) void extract_h(const float* __restrict__ HC,
                                                 float* __restrict__ out) {
    const int idx = blockIdx.x * 256 + threadIdx.x;   // 512*256
    const int b = idx >> 8, d = idx & 255;
    out[idx] = HC[(size_t)b * 2 * DDIM + d];
}

// ---------------------------------------------------------------------------
extern "C" void kernel_launch(void* const* d_in, const int* in_sizes, int n_in,
                              void* d_out, int out_size, void* d_ws, size_t ws_size,
                              hipStream_t stream) {
    (void)in_sizes; (void)n_in; (void)out_size; (void)ws_size;

    const float* X    = (const float*)d_in[0];   // (512,127,256)
    const float* yh   = (const float*)d_in[1];   // (512,127,1)
    const float* W1   = (const float*)d_in[2];   // (256,768)
    const float* b1   = (const float*)d_in[3];   // (256)
    const float* W2   = (const float*)d_in[4];   // (256)   (attn_W2 is (1,256))
    const float* b2   = (const float*)d_in[5];   // (1)
    const float* Wih  = (const float*)d_in[6];   // (1024,1)
    const float* Whh  = (const float*)d_in[7];   // (1024,256)
    const float* bih  = (const float*)d_in[8];   // (1024)
    const float* bhh  = (const float*)d_in[9];   // (1024)
    const float* fcW  = (const float*)d_in[10];  // (1,257)
    const float* fcb  = (const float*)d_in[11];  // (1)
    float* out = (float*)d_out;                  // (512,256)

    // ---- workspace carve-up (all chunk sizes are 256B multiples) ----
    char* p = (char*)d_ws;
    float*  encp  = (float*)p;  p += (size_t)BATCH * TSTEPS * EDIM * 4;   // 66.6 MB
    __bf16* Xb    = (__bf16*)p; p += (size_t)BATCH * TSTEPS * EDIM * 2;   // 33.3 MB
    __bf16* W1e_b = (__bf16*)p; p += (size_t)EDIM * EDIM * 2;             // W1_e   (256x256)
    __bf16* W1hc_b= (__bf16*)p; p += (size_t)EDIM * 2 * DDIM * 2;         // W1_hc  (256x512)
    __bf16* Whh_b = (__bf16*)p; p += (size_t)4 * DDIM * DDIM * 2;         // Whh    (1024x256)
    float*  HC    = (float*)p;  p += (size_t)BATCH * 2 * DDIM * 4;        // [h|c]  fp32
    __bf16* HCb   = (__bf16*)p; p += (size_t)BATCH * 2 * DDIM * 2;        // [h|c]  bf16
    float*  hcp   = (float*)p;  p += (size_t)BATCH * EDIM * 4;            // (512x256)
    float*  gates = (float*)p;  p += (size_t)BATCH * 4 * DDIM * 4;        // (512x1024)
    float*  yt    = (float*)p;  p += (size_t)BATCH * 4;                   // (512)

    // h0 = c0 = 0 every call (ws is not re-poisoned between replays)
    hipMemsetAsync(HC, 0, (size_t)BATCH * 2 * DDIM * sizeof(float), stream);
    hipMemsetAsync(HCb, 0, (size_t)BATCH * 2 * DDIM * sizeof(__bf16), stream);

    // ---- one-time conversions to bf16 ----
    cvt_bf16_flat<<<(BATCH * TSTEPS * EDIM + 255) / 256, 256, 0, stream>>>(
        Xb, X, BATCH * TSTEPS * EDIM);
    cvt_bf16_strided<<<(EDIM * EDIM + 255) / 256, 256, 0, stream>>>(
        W1e_b, W1 + 2 * DDIM, EDIM, EDIM, 2 * DDIM + EDIM);      // W1_e = W1[:,512:768]
    cvt_bf16_strided<<<(EDIM * 2 * DDIM + 255) / 256, 256, 0, stream>>>(
        W1hc_b, W1, EDIM, 2 * DDIM, 2 * DDIM + EDIM);            // W1_hc = W1[:,0:512]
    cvt_bf16_strided<<<(4 * DDIM * DDIM + 255) / 256, 256, 0, stream>>>(
        Whh_b, Whh, 4 * DDIM, DDIM, DDIM);

    // ---- enc_proj precompute (attn_b1 folded in): 65024x256x256 ----
    {
        const int jobs = (BATCH * TSTEPS / 16) * (EDIM / 64);    // 4064*4 = 16256
        gemm_encproj<<<jobs / 8, 256, 0, stream>>>(Xb, W1e_b, b1, encp);
    }

    // ---- 127 sequential decoder steps ----
    for (int step = 0; step < TSTEPS; ++step) {
        gemm_hcproj<<<16, 256, 0, stream>>>(HCb, W1hc_b, hcp);            // 512x256x512
        attn_context<<<BATCH, 256, 0, stream>>>(hcp, encp, X, W2, b2, fcW, fcb,
                                                yh, step, yt);
        gemm_gates<<<64, 256, 0, stream>>>(HCb, Whh_b, yt, Wih, bih, bhh, gates); // 512x1024x256
        lstm_pointwise<<<BATCH * DDIM / 256, 256, 0, stream>>>(gates, HC, HCb);
    }

    // ---- emit final h ----
    extract_h<<<BATCH * DDIM / 256, 256, 0, stream>>>(HC, out);
}
